// Net_55207509623440
// MI455X (gfx1250) — matrix operand order repaired and checked by reading, not compiled, not run on hardware
//
#include <hip/hip_runtime.h>
#include <hip/hip_bf16.h>
#include <math.h>

#define N_NODES   40000
#define N_EDGES   1280000
#define D_FEAT    1024
#define HIDDEN    64
#define NCLS      40

typedef __attribute__((ext_vector_type(16))) _Float16 v16h;
typedef __attribute__((ext_vector_type(8)))  float    v8f;

// ---------------------------------------------------------------- degrees
__global__ __launch_bounds__(256) void deg_init_k(float* __restrict__ deg) {
    int i = blockIdx.x * 256 + threadIdx.x;
    if (i < N_NODES) deg[i] = 1.0f;                 // self-loop contribution
}

__global__ __launch_bounds__(256) void deg_count_k(const int* __restrict__ col,
                                                   float* __restrict__ deg) {
    int e = blockIdx.x * 256 + threadIdx.x;
    if (e < N_EDGES) atomicAdd(&deg[col[e]], 1.0f);
}

__global__ __launch_bounds__(256) void dinv_k(float* __restrict__ deg) {
    int i = blockIdx.x * 256 + threadIdx.x;
    if (i < N_NODES) {
        float d = deg[i];
        deg[i] = (d > 0.0f) ? rsqrtf(d) : 0.0f;
    }
}

// ---------------------------------------------------------------- GEMM1: h1 = x @ W1  (f16 WMMA, f32 accum)
// Block: 256 threads = 8 waves; each wave -> one 16x16 tile; block tile = 32x64.
#define KSLAB 128
#define SPAD  8
#define SSTR  (KSLAB + SPAD)   // 136 halves -> 272B row stride, lanes land on distinct banks

__global__ __launch_bounds__(256) void gemm1_k(const float* __restrict__ x,
                                               const float* __restrict__ W1,
                                               float* __restrict__ h1) {
    __shared__ _Float16 sW[HIDDEN * SSTR];          // 64*136*2 = 17408 B

    const int tid    = threadIdx.x;
    const int wave   = tid >> 5;
    const int lane   = tid & 31;
    const int halfid = lane >> 4;                   // 0 | 1
    const int ln     = lane & 15;

    const int m0 = blockIdx.x * 32 + (wave >> 2) * 16;   // 40000/32 = 1250 blocks
    const int n0 = (wave & 3) * 16;

    v8f acc = {};
    const float* xrow = x + (size_t)(m0 + ln) * D_FEAT + halfid * 8;

    for (int ks = 0; ks < D_FEAT; ks += KSLAB) {
        __syncthreads();
        // stage W1[ks..ks+127][0..63] transposed into LDS as half: sW[n][k]
        for (int idx = tid; idx < KSLAB * HIDDEN; idx += 256) {
            int k = idx >> 6;          // 0..127
            int n = idx & 63;          // 0..63  (coalesced global read)
            sW[n * SSTR + k] = (_Float16)W1[(size_t)(ks + k) * HIDDEN + n];
        }
        __syncthreads();

        const _Float16* brow = sW + (n0 + ln) * SSTR + halfid * 8;
        for (int kb = 0; kb < KSLAB; kb += 32) {
            v16h a, b;
            #pragma unroll
            for (int e = 0; e < 8; ++e) {
                // A 16x32 f16 layout: lanes0-15 K=0..7,16..23 ; lanes16-31 K=8..15,24..31
                a[e]     = (_Float16)xrow[ks + kb + e];
                a[e + 8] = (_Float16)xrow[ks + kb + 16 + e];
            }
            #pragma unroll
            for (int e = 0; e < 16; ++e) {
                // B 32x16 f16: lanes0-15 K=0..7,16..23 ; lanes16-31 K=8..15,24..31
                b[e] = brow[kb + e + ((e >> 3) << 3)];
            }
            acc = __builtin_amdgcn_wmma_f32_16x16x32_f16(false, a, false, b,
                                                         (short)0, acc, false, false);
        }
    }

    #pragma unroll
    for (int r = 0; r < 8; ++r)        // D VGPR r -> row r + 8*halfid, col ln
        h1[(size_t)(m0 + r + 8 * halfid) * HIDDEN + n0 + ln] = acc[r];
}

// ---------------------------------------------------------------- layer-1 aggregation
__global__ __launch_bounds__(256) void agg1_init_k(const float* __restrict__ h1,
                                                   const float* __restrict__ dinv,
                                                   const float* __restrict__ b1,
                                                   float* __restrict__ a1) {
    int idx = blockIdx.x * 256 + threadIdx.x;       // node*64 + f
    if (idx < N_NODES * HIDDEN) {
        int node = idx >> 6, f = idx & 63;
        float d = dinv[node];
        a1[idx] = b1[f] + h1[idx] * d * d;          // self-loop + bias
    }
}

__global__ __launch_bounds__(256) void agg1_scatter_k(const int* __restrict__ row,
                                                      const int* __restrict__ col,
                                                      const float* __restrict__ h1,
                                                      const float* __restrict__ dinv,
                                                      float* __restrict__ a1) {
    unsigned gid = blockIdx.x * 256u + threadIdx.x; // 64 lanes per edge
    unsigned e = gid >> 6, f = gid & 63;
    if (e < N_EDGES) {
        int r = row[e], c = col[e];
        float nrm = dinv[r] * dinv[c];
        atomicAdd(&a1[(size_t)c * HIDDEN + f], h1[(size_t)r * HIDDEN + f] * nrm);
    }
}

__global__ __launch_bounds__(256) void relu_k(float* __restrict__ a1) {
    int idx = blockIdx.x * 256 + threadIdx.x;
    if (idx < N_NODES * HIDDEN) a1[idx] = fmaxf(a1[idx], 0.0f);
}

// ---------------------------------------------------------------- GEMM2: h2 = a1 @ W2  (N padded 40->48)
#define S2STR (HIDDEN + SPAD)   // 72

__global__ __launch_bounds__(96) void gemm2_k(const float* __restrict__ a1,
                                              const float* __restrict__ W2,
                                              float* __restrict__ h2) {
    __shared__ _Float16 sW[48 * S2STR];

    const int tid = threadIdx.x;                    // 96 threads = 3 waves
    for (int idx = tid; idx < 48 * HIDDEN; idx += 96) {
        int n = idx >> 6, k = idx & 63;
        float v = (n < NCLS) ? W2[(size_t)k * NCLS + n] : 0.0f;
        sW[n * S2STR + k] = (_Float16)v;
    }
    __syncthreads();

    const int wave   = tid >> 5;                    // col tile 0..2
    const int lane   = tid & 31;
    const int halfid = lane >> 4;
    const int ln     = lane & 15;
    const int m0 = blockIdx.x * 16;                 // 2500 blocks
    const int n0 = wave * 16;

    v8f acc = {};
    const float*    arow = a1 + (size_t)(m0 + ln) * HIDDEN + halfid * 8;
    const _Float16* brow = sW + (n0 + ln) * S2STR + halfid * 8;

    #pragma unroll
    for (int kb = 0; kb < HIDDEN; kb += 32) {
        v16h a, b;
        #pragma unroll
        for (int e = 0; e < 8; ++e) {
            a[e]     = (_Float16)arow[kb + e];
            a[e + 8] = (_Float16)arow[kb + 16 + e];
        }
        #pragma unroll
        for (int e = 0; e < 16; ++e) b[e] = brow[kb + e + ((e >> 3) << 3)];
        acc = __builtin_amdgcn_wmma_f32_16x16x32_f16(false, a, false, b,
                                                     (short)0, acc, false, false);
    }

    const int colg = n0 + ln;
    if (colg < NCLS) {
        #pragma unroll
        for (int r = 0; r < 8; ++r)
            h2[(size_t)(m0 + r + 8 * halfid) * NCLS + colg] = acc[r];
    }
}

// ---------------------------------------------------------------- layer-2 aggregation (into d_out)
__global__ __launch_bounds__(256) void agg2_init_k(const float* __restrict__ h2,
                                                   const float* __restrict__ dinv,
                                                   const float* __restrict__ b2,
                                                   float* __restrict__ a2) {
    int idx = blockIdx.x * 256 + threadIdx.x;       // node*40 + f
    if (idx < N_NODES * NCLS) {
        int node = idx / NCLS, f = idx - node * NCLS;
        float d = dinv[node];
        a2[idx] = b2[f] + h2[idx] * d * d;
    }
}

__global__ __launch_bounds__(256) void agg2_scatter_k(const int* __restrict__ row,
                                                      const int* __restrict__ col,
                                                      const float* __restrict__ h2,
                                                      const float* __restrict__ dinv,
                                                      float* __restrict__ a2) {
    unsigned gid = blockIdx.x * 256u + threadIdx.x; // E*40 threads
    if (gid < (unsigned)N_EDGES * NCLS) {
        unsigned e = gid / NCLS, f = gid - e * NCLS;
        int r = row[e], c = col[e];
        float nrm = dinv[r] * dinv[c];
        atomicAdd(&a2[(size_t)c * NCLS + f], h2[(size_t)r * NCLS + f] * nrm);
    }
}

__global__ __launch_bounds__(256) void logsoftmax_k(float* __restrict__ out) {
    int node = blockIdx.x * 256 + threadIdx.x;
    if (node < N_NODES) {
        float* p = out + (size_t)node * NCLS;
        float mx = -INFINITY;
        #pragma unroll
        for (int i = 0; i < NCLS; ++i) mx = fmaxf(mx, p[i]);
        float s = 0.0f;
        #pragma unroll
        for (int i = 0; i < NCLS; ++i) s += __expf(p[i] - mx);
        float lse = mx + __logf(s);
        #pragma unroll
        for (int i = 0; i < NCLS; ++i) p[i] = p[i] - lse;   // in place, thread-private row
    }
}

// ---------------------------------------------------------------- launch
extern "C" void kernel_launch(void* const* d_in, const int* in_sizes, int n_in,
                              void* d_out, int out_size, void* d_ws, size_t ws_size,
                              hipStream_t stream) {
    const float* x   = (const float*)d_in[0];
    const int*   row = (const int*)d_in[1];           // edge_index[0] = source
    const int*   col = row + N_EDGES;                 // edge_index[1] = target
    const float* W1  = (const float*)d_in[2];
    const float* b1  = (const float*)d_in[3];
    const float* W2  = (const float*)d_in[4];
    const float* b2  = (const float*)d_in[5];
    float*       out = (float*)d_out;

    float* ws   = (float*)d_ws;
    float* dinv = ws;                                 // 40000
    float* h1   = ws + N_NODES;                       // 40000*64
    float* a1   = h1 + (size_t)N_NODES * HIDDEN;      // 40000*64
    float* h2   = h1;                                 // reuse (h1 dead after scatter1)
    float* a2   = out;                                // accumulate layer-2 directly in d_out

    // degrees -> dinv
    deg_init_k <<<(N_NODES + 255) / 256, 256, 0, stream>>>(dinv);
    deg_count_k<<<(N_EDGES + 255) / 256, 256, 0, stream>>>(col, dinv);
    dinv_k     <<<(N_NODES + 255) / 256, 256, 0, stream>>>(dinv);

    // layer 1
    gemm1_k<<<N_NODES / 32, 256, 0, stream>>>(x, W1, h1);
    agg1_init_k   <<<(N_NODES * HIDDEN + 255) / 256, 256, 0, stream>>>(h1, dinv, b1, a1);
    agg1_scatter_k<<<((size_t)N_EDGES * 64 + 255) / 256, 256, 0, stream>>>(row, col, h1, dinv, a1);
    relu_k        <<<(N_NODES * HIDDEN + 255) / 256, 256, 0, stream>>>(a1);

    // layer 2
    gemm2_k<<<N_NODES / 16, 96, 0, stream>>>(a1, W2, h2);
    agg2_init_k   <<<(N_NODES * NCLS + 255) / 256, 256, 0, stream>>>(h2, dinv, b2, a2);
    agg2_scatter_k<<<((size_t)N_EDGES * NCLS + 255) / 256, 256, 0, stream>>>(row, col, h2, dinv, a2);

    // output
    logsoftmax_k<<<(N_NODES + 255) / 256, 256, 0, stream>>>(out);
}
